// AttentionNet_33732673143065
// MI455X (gfx1250) — matrix operand order repaired and checked by reading, not compiled, not run on hardware
//
#include <hip/hip_runtime.h>
#include <hip/hip_bf16.h>

typedef __attribute__((ext_vector_type(16))) __bf16          v16bf;
typedef __attribute__((ext_vector_type(2)))  __bf16          v2bf;
typedef __attribute__((ext_vector_type(8)))  float           v8f;
typedef __attribute__((ext_vector_type(8)))  unsigned short  us8;
typedef __attribute__((ext_vector_type(16))) unsigned short  us16;

#define BSZ  64
#define SSZ  2048
#define HSZ  512
#define RSZ  256
#define NEGV (-1.0e9f)
#define SCH  8            // s-chunks for weighted pooling

__device__ __forceinline__ unsigned short f2bf(float f) {
    unsigned int u = __float_as_uint(f);
    unsigned int r = u + 0x7FFFu + ((u >> 16) & 1u);   // round-to-nearest-even
    return (unsigned short)(r >> 16);
}

// packed fp32x2 -> bf16x2 (hardware V_CVT_PK_BF16_F32 when available)
__device__ __forceinline__ unsigned int f2bf_pk(float lo, float hi) {
#if __has_builtin(__builtin_amdgcn_cvt_pk_bf16_f32)
    v2bf p = __builtin_amdgcn_cvt_pk_bf16_f32(lo, hi);
    return __builtin_bit_cast(unsigned int, p);
#else
    return (unsigned int)f2bf(lo) | ((unsigned int)f2bf(hi) << 16);
#endif
}

__device__ __forceinline__ float fast_tanh(float x) {
#if __has_builtin(__builtin_amdgcn_tanhf)
    return __builtin_amdgcn_tanhf(x);
#elif __has_builtin(__builtin_amdgcn_tanh_f32)
    return __builtin_amdgcn_tanh_f32(x);
#else
    return tanhf(x);
#endif
}

// ---------------------------------------------------------------------------
// K1: base[b,h] = rel[b,:]·Wr_w[h,:] + Wr_b[h] + pool[b,:]·Wg_w[h,:] + Wg_b[h] + Wh_b[h]
// ---------------------------------------------------------------------------
__global__ __launch_bounds__(512)
void base_kernel(const float* __restrict__ rel,  const float* __restrict__ pool,
                 const float* __restrict__ Wr_w, const float* __restrict__ Wr_b,
                 const float* __restrict__ Wg_w, const float* __restrict__ Wg_b,
                 const float* __restrict__ Wh_b, float* __restrict__ base) {
    __shared__ float sr[RSZ];
    __shared__ float sp[HSZ];
    const int b = blockIdx.x;
    const int h = threadIdx.x;
    if (h < RSZ) sr[h] = rel[b * RSZ + h];
    sp[h] = pool[b * HSZ + h];
    __syncthreads();
    float acc = Wr_b[h] + Wg_b[h] + Wh_b[h];
    const float* wr = Wr_w + (size_t)h * RSZ;
    #pragma unroll 4
    for (int r = 0; r < RSZ; ++r) acc += sr[r] * wr[r];
    const float* wg = Wg_w + (size_t)h * HSZ;
    #pragma unroll 4
    for (int j = 0; j < HSZ; ++j) acc += sp[j] * wg[j];
    base[b * HSZ + h] = acc;
}

// ---------------------------------------------------------------------------
// K1b: pre-convert Wh_w (H_out x H_in, row major) into bf16 WMMA B-fragments.
// B (32x16) lane layout: lane = half*16 + n; lane holds K = half*16 + 0..15
// element B[k][n] = Wh_w[ntile*16+n][ks*32+k]
// ---------------------------------------------------------------------------
__global__ __launch_bounds__(256)
void whb_kernel(const float* __restrict__ Wh_w, unsigned short* __restrict__ WhB) {
    int idx  = blockIdx.x * 256 + threadIdx.x;   // 16 ks * 32 nt * 32 lanes = 16384
    int lane = idx & 31;
    int nt   = (idx >> 5) & 31;
    int ks   = idx >> 10;
    int row  = nt * 16 + (lane & 15);
    int col0 = ks * 32 + ((lane >> 4) * 16);
    unsigned int* dst = (unsigned int*)(WhB + ((size_t)(ks * 32 + nt) * 32 + lane) * 16);
    const float* src = Wh_w + (size_t)row * HSZ + col0;
    #pragma unroll
    for (int j = 0; j < 8; ++j) dst[j] = f2bf_pk(src[2 * j], src[2 * j + 1]);
}

// ---------------------------------------------------------------------------
// K2: main fused kernel. Per workgroup: 64 rows of (b,s), full N=512.
//   sent_row = sent_h_row @ Wh^T  (bf16 WMMA, f32 accum)
//   w_raw    = alpha · tanh(sent_row + base[b,:]) + alpha_b, masked
// 8 waves; wave w owns N columns [w*64, w*64+64) for all 4 M-tiles.
// ---------------------------------------------------------------------------
__global__ __launch_bounds__(256)
void fused_main_kernel(const float* __restrict__ sent_h,
                       const unsigned short* __restrict__ WhB,
                       const float* __restrict__ base,
                       const float* __restrict__ alpha_w,
                       const float* __restrict__ alpha_b,
                       const int*   __restrict__ mask,
                       float* __restrict__ wraw) {
    __shared__ unsigned short Alds[64 * 40];   // 64 rows x 32 K, stride 40 (16B aligned)
    __shared__ float part[8][64];

    const int tid   = threadIdx.x;
    const int lane  = tid & 31;
    const int wave  = tid >> 5;
    const int half  = lane >> 4;       // 0: lanes 0-15, 1: lanes 16-31
    const int l15   = lane & 15;
    const int mbase = blockIdx.x * 64; // S%64==0 -> never crosses batch boundary
    const int b     = mbase / SSZ;
    const int s0    = mbase - b * SSZ;
    const int kbaseA = half * 16;      // A-frag K runs: {kbaseA..+15}

    v8f acc[4][4] = {};

    const int ldr = tid >> 2;          // coop-load row 0..63
    const int ldc = (tid & 3) * 8;     // coop-load col {0,8,16,24}
    const float* ldsrc = sent_h + (size_t)(mbase + ldr) * HSZ + ldc;

    for (int ks = 0; ks < 16; ++ks) {
        if (ks + 1 < 16)
            __builtin_prefetch(ldsrc + (ks + 1) * 32, 0, 1);
        __syncthreads();
        // ---- cooperative stage: 64x32 fp32 -> bf16 (packed cvt) -> LDS ----
        {
            const float4 f0 = *(const float4*)(ldsrc + ks * 32);
            const float4 f1 = *(const float4*)(ldsrc + ks * 32 + 4);
            typedef __attribute__((ext_vector_type(4))) unsigned int uint4v;
            uint4v v;
            v[0] = f2bf_pk(f0.x, f0.y);
            v[1] = f2bf_pk(f0.z, f0.w);
            v[2] = f2bf_pk(f1.x, f1.y);
            v[3] = f2bf_pk(f1.z, f1.w);
            *(uint4v*)&Alds[ldr * 40 + ldc] = v;
        }
        __syncthreads();
        // ---- B fragments from pre-converted global (L2-resident) ----
        v16bf bfrag[4];
        #pragma unroll
        for (int nt = 0; nt < 4; ++nt) {
            const int ntg = wave * 4 + nt;
            const us16* p = (const us16*)(WhB + ((size_t)(ks * 32 + ntg) * 32 + lane) * 16);
            bfrag[nt] = __builtin_bit_cast(v16bf, *p);
        }
        // ---- A fragments + 16 WMMAs ----
        #pragma unroll
        for (int mt = 0; mt < 4; ++mt) {
            const int row = mt * 16 + l15;
            us8 lo = *(const us8*)&Alds[row * 40 + kbaseA];
            us8 hi = *(const us8*)&Alds[row * 40 + kbaseA + 8];
            us16 u = __builtin_shufflevector(lo, hi, 0,1,2,3,4,5,6,7,8,9,10,11,12,13,14,15);
            v16bf afrag = __builtin_bit_cast(v16bf, u);
            #pragma unroll
            for (int nt = 0; nt < 4; ++nt) {
                acc[mt][nt] = __builtin_amdgcn_wmma_f32_16x16x32_bf16(
                    false, afrag, false, bfrag[nt], (short)0, acc[mt][nt], false, false);
            }
        }
    }

    // ---- epilogue: +base, tanh, alpha-dot, cross-lane & cross-wave reduce ----
    const int wbase = wave * 64;
    float basereg[4], alphareg[4];
    #pragma unroll
    for (int nt = 0; nt < 4; ++nt) {
        const int col = wbase + nt * 16 + l15;
        basereg[nt]  = base[b * HSZ + col];
        alphareg[nt] = alpha_w[col];
    }
    #pragma unroll
    for (int mt = 0; mt < 4; ++mt) {
        #pragma unroll
        for (int i = 0; i < 8; ++i) {
            float sacc = 0.0f;
            #pragma unroll
            for (int nt = 0; nt < 4; ++nt) {
                float val = acc[mt][nt][i] + basereg[nt];
                sacc += alphareg[nt] * fast_tanh(val);
            }
            #pragma unroll
            for (int off = 1; off < 16; off <<= 1)
                sacc += __shfl_xor(sacc, off, 32);
            const int rowl = mt * 16 + i + half * 8;   // C layout: M = i + 8*half
            if (l15 == 0) part[wave][rowl] = sacc;
        }
    }
    __syncthreads();
    if (tid < 64) {
        float tot = alpha_b[0];
        #pragma unroll
        for (int w = 0; w < 8; ++w) tot += part[w][tid];
        const int s = s0 + tid;
        if (mask[b * SSZ + s] == 0) tot = NEGV;
        wraw[(size_t)b * SSZ + s] = tot;
    }
}

// ---------------------------------------------------------------------------
// K3: in-place softmax over S per batch (raw scores already in d_out)
// ---------------------------------------------------------------------------
__global__ __launch_bounds__(256)
void softmax_kernel(float* __restrict__ wbuf) {
    __shared__ float red[256];
    const int b = blockIdx.x;
    const int t = threadIdx.x;
    float* w = wbuf + (size_t)b * SSZ;
    float v[SSZ / 256];
    float mx = -3.4e38f;
    #pragma unroll
    for (int k = 0; k < SSZ / 256; ++k) { v[k] = w[t + k * 256]; mx = fmaxf(mx, v[k]); }
    red[t] = mx; __syncthreads();
    for (int o = 128; o > 0; o >>= 1) { if (t < o) red[t] = fmaxf(red[t], red[t + o]); __syncthreads(); }
    mx = red[0]; __syncthreads();
    float sum = 0.0f;
    #pragma unroll
    for (int k = 0; k < SSZ / 256; ++k) { v[k] = __expf(v[k] - mx); sum += v[k]; }
    red[t] = sum; __syncthreads();
    for (int o = 128; o > 0; o >>= 1) { if (t < o) red[t] += red[t + o]; __syncthreads(); }
    const float inv = 1.0f / red[0];
    #pragma unroll
    for (int k = 0; k < SSZ / 256; ++k) w[t + k * 256] = v[k] * inv;
}

// ---------------------------------------------------------------------------
// K4: weighted pooling of RAW input: vpart[ch,b,h] = sum_{s in chunk} w[b,s]*sent_h[b,s,h]
// ---------------------------------------------------------------------------
__global__ __launch_bounds__(256)
void wpool_kernel(const float* __restrict__ sent_h, const float* __restrict__ wsm,
                  float* __restrict__ vpart) {
    const int blk = blockIdx.x;           // 64*8*2 blocks
    const int ht  = blk & 1;
    const int ch  = (blk >> 1) & 7;
    const int b   = blk >> 4;
    const int h   = ht * 256 + threadIdx.x;
    const int sbeg = ch * (SSZ / SCH);
    const float* x = sent_h + ((size_t)b * SSZ + sbeg) * HSZ + h;
    const float* wp = wsm + (size_t)b * SSZ + sbeg;
    float acc = 0.0f;
    #pragma unroll 4
    for (int s = 0; s < SSZ / SCH; ++s) acc += wp[s] * x[(size_t)s * HSZ];
    vpart[((size_t)ch * BSZ + b) * HSZ + h] = acc;
}

// ---------------------------------------------------------------------------
// K5: att_res[b,k] = v[b,:]·Wh_w[k,:] + Wh_b[k]   (valid since sum(w)==1)
// ---------------------------------------------------------------------------
__global__ __launch_bounds__(512)
void attres_kernel(const float* __restrict__ vpart, const float* __restrict__ Wh_w,
                   const float* __restrict__ Wh_b, float* __restrict__ out) {
    __shared__ float vt[HSZ];
    const int b = blockIdx.x;
    const int k = threadIdx.x;
    float s = 0.0f;
    #pragma unroll
    for (int ch = 0; ch < SCH; ++ch) s += vpart[((size_t)ch * BSZ + b) * HSZ + k];
    vt[k] = s;
    __syncthreads();
    float acc = Wh_b[k];
    const float* wr = Wh_w + (size_t)k * HSZ;
    #pragma unroll 4
    for (int h = 0; h < HSZ; ++h) acc += vt[h] * wr[h];
    out[(size_t)b * HSZ + k] = acc;
}

// ---------------------------------------------------------------------------
extern "C" void kernel_launch(void* const* d_in, const int* in_sizes, int n_in,
                              void* d_out, int out_size, void* d_ws, size_t ws_size,
                              hipStream_t stream) {
    const float* sent_h  = (const float*)d_in[0];
    const float* rel     = (const float*)d_in[1];
    const float* pool    = (const float*)d_in[2];
    const int*   mask    = (const int*)  d_in[3];
    const float* Wg_w    = (const float*)d_in[4];
    const float* Wg_b    = (const float*)d_in[5];
    const float* Wh_w    = (const float*)d_in[6];
    const float* Wh_b    = (const float*)d_in[7];
    const float* Wr_w    = (const float*)d_in[8];
    const float* Wr_b    = (const float*)d_in[9];
    const float* alpha_w = (const float*)d_in[10];
    const float* alpha_b = (const float*)d_in[11];

    float* out  = (float*)d_out;
    float* wbuf = out + BSZ * HSZ;                 // weight_ region of d_out (raw -> softmax in place)

    // workspace layout
    char* ws = (char*)d_ws;
    float*          base  = (float*)ws;                                   // 128 KB
    unsigned short* WhB   = (unsigned short*)(ws + 131072);               // 512 KB
    float*          vpart = (float*)(ws + 131072 + 524288);               // 1 MB

    base_kernel  <<<BSZ,               512, 0, stream>>>(rel, pool, Wr_w, Wr_b, Wg_w, Wg_b, Wh_b, base);
    whb_kernel   <<<64,                256, 0, stream>>>(Wh_w, WhB);
    fused_main_kernel<<<(BSZ * SSZ)/64,256, 0, stream>>>(sent_h, WhB, base, alpha_w, alpha_b, mask, wbuf);
    softmax_kernel<<<BSZ,              256, 0, stream>>>(wbuf);
    wpool_kernel <<<BSZ * SCH * 2,     256, 0, stream>>>(sent_h, wbuf, vpart);
    attres_kernel<<<BSZ,               512, 0, stream>>>(vpart, Wh_w, Wh_b, out);
}
